// SPN_23416161698328
// MI455X (gfx1250) — matrix operand order repaired and checked by reading, not compiled, not run on hardware
//
#include <hip/hip_runtime.h>

// ---------------------------------------------------------------------------
// SPN pipeline for MI455X (gfx1250, wave32).
// Convs: implicit GEMM on v_wmma_f32_16x16x32_f16 (f16 in, fp32 accumulate).
// Scans: LDS double-buffered per-(b,c)-plane sequential propagation.
// Fixed problem shapes (from setup_inputs):
//   featureMap (8,19,128,256)  guidance (8,256,64,128)
//   conv1: 4x4 s2 p1 -> f (8,32,64,128)
//   conv2: 3x3 s1 p1 -> g (8,384,64,128)
//   conv3: 3x3 s1 p1 on upsampled (8,32,128,256) -> (8,64,128,256), ReLU
//   conv4: 3x3 s1 p1 -> (8,19,128,256) fp32 to d_out
// Workspace use ~190 MB (with aliasing of dead buffers).
// ---------------------------------------------------------------------------

typedef __attribute__((ext_vector_type(16))) _Float16 v16h;
typedef __attribute__((ext_vector_type(8)))  float    v8f;

// ---------------------------- elementwise f32 -> f16 ------------------------
__global__ void f32_to_f16_kernel(const float* __restrict__ in,
                                  _Float16* __restrict__ out, size_t n) {
  size_t t = (size_t)blockIdx.x * blockDim.x + threadIdx.x;
  if (t < n) out[t] = (_Float16)in[t];
}

// ------------------- pack weights into WMMA B-fragment layout ---------------
// Source: OIHW fp32, flattened w[n*K + k] with k = cin*R*S + r*S + s.
// Dest:   wp[(((kb*NT + nt)*32 + lane)*16 + e],
//         k = kb*32 + e + 16*(lane>>4), n = nt*16 + (lane&15); zero padded.
__global__ void pack_w_kernel(const float* __restrict__ w, _Float16* __restrict__ wp,
                              int Cout, int K, int Kpad, int NPad) {
  size_t t = (size_t)blockIdx.x * blockDim.x + threadIdx.x;
  size_t total = (size_t)Kpad * NPad;
  if (t >= total) return;
  int e    = (int)(t & 15);
  int lane = (int)((t >> 4) & 31);
  int NT   = NPad >> 4;
  int nt   = (int)((t >> 9) % NT);
  int kb   = (int)((t >> 9) / NT);
  int k = kb * 32 + e + 16 * (lane >> 4);
  int n = nt * 16 + (lane & 15);
  float v = 0.0f;
  if (k < K && n < Cout) v = w[(size_t)n * K + k];
  wp[t] = (_Float16)v;
}

// --------------------------- implicit-GEMM conv -----------------------------
// One wave computes a 16(M) x 64(N) output tile: one A fragment shared by
// 4 WMMA accumulators. A gathered from x (f16, NCHW) with implicit im2col.
template <int R, int S, int STRIDE, int PAD, bool RELU, bool OUT_F16>
__global__ void __launch_bounds__(256)
conv_wmma_kernel(const _Float16* __restrict__ x, const _Float16* __restrict__ wp,
                 void* __restrict__ outp,
                 int Bn, int Cin, int Hin, int Win,
                 int Cout, int Hout, int Wout, int NPad) {
  constexpr int RS = R * S;
  const int lane   = threadIdx.x & 31;
  const int laneLo = lane & 15;
  const int laneHi = lane >> 4;

  const int K      = Cin * RS;
  const int M      = Bn * Hout * Wout;
  const int Mtiles = (M + 15) >> 4;
  const int N64    = NPad >> 6;
  const int NT     = NPad >> 4;

  const int wave = blockIdx.x * (blockDim.x >> 5) + (threadIdx.x >> 5);
  if (wave >= Mtiles * N64) return;          // wave-uniform: EXEC stays full
  const int mTile = wave % Mtiles;
  const int n64   = wave / Mtiles;
  const int HWo   = Hout * Wout;

  // A-row coordinates for this lane (row = laneLo of the 16-row tile)
  const int mA = mTile * 16 + laneLo;
  const bool mAvalid = (mA < M);
  int bA = 0, oyA = 0, oxA = 0;
  if (mAvalid) {
    bA = mA / HWo;
    int r = mA - bA * HWo;
    oyA = r / Wout;
    oxA = r - oyA * Wout;
  }
  const int iy0 = oyA * STRIDE - PAD;
  const int ix0 = oxA * STRIDE - PAD;

  const v8f zero8 = {0.f, 0.f, 0.f, 0.f, 0.f, 0.f, 0.f, 0.f};
  v8f acc[4] = {zero8, zero8, zero8, zero8};

  for (int k0 = 0; k0 < K; k0 += 32) {
    // ---- A fragment: contiguous k order (same as packed B) ----
    // element e of lane: K = k0 + e + 16*laneHi, M row = laneLo
    v16h a;
#pragma unroll
    for (int e = 0; e < 16; ++e) {
      int ke = k0 + e + 16 * laneHi;
      _Float16 v = (_Float16)0.f;
      if (mAvalid && ke < K) {
        int cin = ke / RS;
        int rs  = ke - cin * RS;
        int r   = rs / S;
        int s   = rs - r * S;
        int iy  = iy0 + r;
        int ix  = ix0 + s;
        if ((unsigned)iy < (unsigned)Hin && (unsigned)ix < (unsigned)Win)
          v = x[(((size_t)bA * Cin + cin) * Hin + iy) * (size_t)Win + ix];
      }
      a[e] = v;
    }
    // ---- 4 x (B fragment load + WMMA) ----
    const size_t kbBase = (size_t)(k0 >> 5) * NT;
#pragma unroll
    for (int t = 0; t < 4; ++t) {
      int ntile = (n64 << 2) + t;
      const v16h bfrag =
          *(const v16h*)(wp + ((kbBase + ntile) * 32 + lane) * 16);
      acc[t] = __builtin_amdgcn_wmma_f32_16x16x32_f16(
          false, a, false, bfrag, (short)0, acc[t], false, false);
    }
  }

  // ---- store: C/D layout row = r + 8*laneHi, col = laneLo ----
  int mrB[8], mrOy[8], mrOx[8];
  bool mrV[8];
#pragma unroll
  for (int r = 0; r < 8; ++r) {
    int m = mTile * 16 + r + 8 * laneHi;
    mrV[r] = (m < M);
    int mm = mrV[r] ? m : 0;
    mrB[r] = mm / HWo;
    int rem = mm - mrB[r] * HWo;
    mrOy[r] = rem / Wout;
    mrOx[r] = rem - mrOy[r] * Wout;
  }
#pragma unroll
  for (int t = 0; t < 4; ++t) {
    int n = (n64 << 6) + t * 16 + laneLo;
    if (n >= Cout) continue;
#pragma unroll
    for (int r = 0; r < 8; ++r) {
      if (!mrV[r]) continue;
      float v = acc[t][r];
      if (RELU) v = fmaxf(v, 0.f);
      size_t oidx =
          (((size_t)mrB[r] * Cout + n) * Hout + mrOy[r]) * (size_t)Wout + mrOx[r];
      if (OUT_F16) ((_Float16*)outp)[oidx] = (_Float16)v;
      else         ((float*)outp)[oidx]    = v;
    }
  }
}

// ------------------------------ gate normalization --------------------------
// g (8,384,64,128) fp32, in place. Replicates reference exactly:
//   gates 0-2  /= sumabs(0..2)   if >= 1
//   gates 3-5  /= sumabs(3..5)   if >= 1
//   gates 6-8  /= sumabs(6..11)  if >= 1   (six-gate sum!)
//   gates 9-11 /= sumabs(9..11)  if >= 1   (from ORIGINAL values)
__global__ void gnorm_kernel(float* __restrict__ g) {
  const int HW = 64 * 128;
  size_t t = (size_t)blockIdx.x * blockDim.x + threadIdx.x;
  const size_t total = (size_t)8 * 32 * HW;
  if (t >= total) return;
  int b   = (int)(t / ((size_t)32 * HW));
  int rem = (int)(t - (size_t)b * 32 * HW);
  int cc  = rem / HW;
  int p   = rem - cc * HW;
  size_t base = ((size_t)b * 384 + cc) * HW + p;
  const size_t gs = (size_t)32 * HW;
  float v[12];
#pragma unroll
  for (int i = 0; i < 12; ++i) v[i] = g[base + i * gs];
  float s1 = fabsf(v[0]) + fabsf(v[1]) + fabsf(v[2]);
  float s2 = fabsf(v[3]) + fabsf(v[4]) + fabsf(v[5]);
  float s3 = fabsf(v[6]) + fabsf(v[7]) + fabsf(v[8]) +
             fabsf(v[9]) + fabsf(v[10]) + fabsf(v[11]);
  float s4 = fabsf(v[9]) + fabsf(v[10]) + fabsf(v[11]);
  if (s1 >= 1.f) { v[0] /= s1; v[1] /= s1; v[2] /= s1; }
  if (s2 >= 1.f) { v[3] /= s2; v[4] /= s2; v[5] /= s2; }
  if (s3 >= 1.f) { v[6] /= s3; v[7] /= s3; v[8] /= s3; }
  if (s4 >= 1.f) { v[9] /= s4; v[10] /= s4; v[11] /= s4; }
#pragma unroll
  for (int i = 0; i < 12; ++i) g[base + i * gs] = v[i];
}

// ------------------------------ propagation scans ---------------------------
// Horizontal: scan over x (W=128), state over y (H=64). One block per (b,c).
template <bool REV>
__global__ void hprop_kernel(const float* __restrict__ f, const float* __restrict__ g,
                             int gateCh, float* __restrict__ o) {
  __shared__ float hb[2][64];
  const int y = threadIdx.x;                 // 64 threads
  const int b = blockIdx.x >> 5;
  const int c = blockIdx.x & 31;
  const size_t fB  = (((size_t)b * 32 + c) * 64 + y) * 128;
  const size_t g1B = (((size_t)b * 384 + gateCh + c) * 64 + y) * 128;
  const size_t gs  = (size_t)32 * 64 * 128;
  hb[0][y] = 0.f;
  __syncthreads();
  int cur = 0;
  for (int i = 0; i < 128; ++i) {
    int x = REV ? (127 - i) : i;
    float xt  = f[fB + x];
    float g1  = g[g1B + x];
    float g2v = g[g1B + gs + x];
    float g3  = g[g1B + 2 * gs + x];
    float hm = (y > 0)  ? hb[cur][y - 1] : 0.f;
    float hc = hb[cur][y];
    float hp = (y < 63) ? hb[cur][y + 1] : 0.f;
    float hn = (1.f - g1 - g2v - g3) * xt + g1 * hm + g2v * hc + g3 * hp;
    o[fB + x] = hn;
    hb[cur ^ 1][y] = hn;
    __syncthreads();
    cur ^= 1;
  }
}

// Vertical: scan over y (H=64), state over x (W=128). One block per (b,c).
template <bool REV>
__global__ void vprop_kernel(const float* __restrict__ f, const float* __restrict__ g,
                             int gateCh, float* __restrict__ o) {
  __shared__ float hb[2][128];
  const int x = threadIdx.x;                 // 128 threads
  const int b = blockIdx.x >> 5;
  const int c = blockIdx.x & 31;
  const size_t fB  = (((size_t)b * 32 + c) * 64) * 128 + x;
  const size_t g1B = (((size_t)b * 384 + gateCh + c) * 64) * 128 + x;
  const size_t gs  = (size_t)32 * 64 * 128;
  hb[0][x] = 0.f;
  __syncthreads();
  int cur = 0;
  for (int i = 0; i < 64; ++i) {
    int y = REV ? (63 - i) : i;
    size_t row = (size_t)y * 128;
    float xt  = f[fB + row];
    float g1  = g[g1B + row];
    float g2v = g[g1B + gs + row];
    float g3  = g[g1B + 2 * gs + row];
    float hm = (x > 0)   ? hb[cur][x - 1] : 0.f;
    float hc = hb[cur][x];
    float hp = (x < 127) ? hb[cur][x + 1] : 0.f;
    float hn = (1.f - g1 - g2v - g3) * xt + g1 * hm + g2v * hc + g3 * hp;
    o[fB + row] = hn;
    hb[cur ^ 1][x] = hn;
    __syncthreads();
    cur ^= 1;
  }
}

// ------------------- max(4 dirs) + bilinear 2x upsample -> f16 --------------
__global__ void maxup_kernel(const float* __restrict__ o0, const float* __restrict__ o1,
                             const float* __restrict__ o2, const float* __restrict__ o3,
                             _Float16* __restrict__ oup) {
  size_t t = (size_t)blockIdx.x * blockDim.x + threadIdx.x;
  const size_t total = (size_t)8 * 32 * 128 * 256;
  if (t >= total) return;
  int X  = (int)(t & 255);
  int Y  = (int)((t >> 8) & 127);
  int bc = (int)(t >> 15);                   // 0..255 = b*32+c
  float sy = Y * 0.5f - 0.25f;
  float sx = X * 0.5f - 0.25f;
  int y0 = (int)floorf(sy);
  int x0 = (int)floorf(sx);
  float fy = sy - (float)y0;
  float fx = sx - (float)x0;
  int y0c = y0 < 0 ? 0 : (y0 > 63 ? 63 : y0);
  int y1t = y0 + 1;
  int y1c = y1t < 0 ? 0 : (y1t > 63 ? 63 : y1t);
  int x0c = x0 < 0 ? 0 : (x0 > 127 ? 127 : x0);
  int x1t = x0 + 1;
  int x1c = x1t < 0 ? 0 : (x1t > 127 ? 127 : x1t);
  size_t base = (size_t)bc * 64 * 128;
  auto M4 = [&](int yy, int xx) {
    size_t i = base + (size_t)yy * 128 + xx;
    return fmaxf(fmaxf(o0[i], o1[i]), fmaxf(o2[i], o3[i]));
  };
  float v00 = M4(y0c, x0c), v01 = M4(y0c, x1c);
  float v10 = M4(y1c, x0c), v11 = M4(y1c, x1c);
  float v = (1.f - fy) * ((1.f - fx) * v00 + fx * v01) +
            fy * ((1.f - fx) * v10 + fx * v11);
  oup[t] = (_Float16)v;
}

// ---------------------------------------------------------------------------
extern "C" void kernel_launch(void* const* d_in, const int* in_sizes, int n_in,
                              void* d_out, int out_size, void* d_ws, size_t ws_size,
                              hipStream_t stream) {
  (void)in_sizes; (void)n_in; (void)out_size; (void)ws_size;

  const float* fm = (const float*)d_in[0];   // (8,19,128,256)
  const float* gd = (const float*)d_in[1];   // (8,256,64,128)
  const float* W1 = (const float*)d_in[2];   // (32,19,4,4)
  const float* W2 = (const float*)d_in[3];   // (384,256,3,3)
  const float* W3 = (const float*)d_in[4];   // (64,32,3,3)
  const float* W4 = (const float*)d_in[5];   // (19,64,3,3)

  char* ws = (char*)d_ws;
  size_t off = 0;
  auto take = [&](size_t bytes) -> char* {
    char* p = ws + off;
    off = (off + bytes + 255) & ~(size_t)255;
    return p;
  };

  const size_t FM_N  = (size_t)8 * 19 * 128 * 256;  // 4,980,736
  const size_t GD_N  = (size_t)8 * 256 * 64 * 128;  // 16,777,216
  const size_t F_N   = (size_t)8 * 32 * 64 * 128;   // 2,097,152
  const size_t G_N   = (size_t)8 * 384 * 64 * 128;  // 25,165,824
  const size_t OUP_N = (size_t)8 * 32 * 128 * 256;  // 8,388,608

  _Float16* fm16 = (_Float16*)take(FM_N * 2);
  _Float16* gd16 = (_Float16*)take(GD_N * 2);
  _Float16* w1p  = (_Float16*)take((size_t)320  * 64  * 2);  // K=304->320, N->64
  _Float16* w2p  = (_Float16*)take((size_t)2304 * 384 * 2);
  _Float16* w3p  = (_Float16*)take((size_t)288  * 64  * 2);
  _Float16* w4p  = (_Float16*)take((size_t)576  * 64  * 2);
  float* fbuf = (float*)take(F_N * 4);
  float* gbuf = (float*)take(G_N * 4);
  float* o0   = (float*)take(F_N * 4);
  float* o1   = (float*)take(F_N * 4);
  float* o2   = (float*)take(F_N * 4);
  float* o3   = (float*)take(F_N * 4);
  // Aliases of dead buffers:
  _Float16* oup16 = (_Float16*)gbuf;   // gates dead after propagation
  _Float16* h1_16 = (_Float16*)gd16;   // guidance-f16 dead after conv2 (same size)

  // --- input / weight preparation ---
  f32_to_f16_kernel<<<(unsigned)((FM_N + 255) / 256), 256, 0, stream>>>(fm, fm16, FM_N);
  f32_to_f16_kernel<<<(unsigned)((GD_N + 255) / 256), 256, 0, stream>>>(gd, gd16, GD_N);
  pack_w_kernel<<<(320 * 64 + 255) / 256, 256, 0, stream>>>(W1, w1p, 32, 304, 320, 64);
  pack_w_kernel<<<(2304 * 384 + 255) / 256, 256, 0, stream>>>(W2, w2p, 384, 2304, 2304, 384);
  pack_w_kernel<<<(288 * 64 + 255) / 256, 256, 0, stream>>>(W3, w3p, 64, 288, 288, 64);
  pack_w_kernel<<<(576 * 64 + 255) / 256, 256, 0, stream>>>(W4, w4p, 19, 576, 576, 64);

  // --- conv1: f = conv(featureMap, W1, s2, p1) -> (8,32,64,128) fp32 ---
  // M = 65536 -> 4096 M-tiles x 1 N-group = 4096 waves -> 512 blocks of 8 waves
  conv_wmma_kernel<4, 4, 2, 1, false, false><<<512, 256, 0, stream>>>(
      fm16, w1p, fbuf, 8, 19, 128, 256, 32, 64, 128, 64);

  // --- conv2: g = conv(guidance, W2, s1, p1) -> (8,384,64,128) fp32 ---
  // 4096 M-tiles x 6 N-groups = 24576 waves -> 3072 blocks
  conv_wmma_kernel<3, 3, 1, 1, false, false><<<3072, 256, 0, stream>>>(
      gd16, w2p, gbuf, 8, 256, 64, 128, 384, 64, 128, 384);

  // --- gate normalization (in place) ---
  gnorm_kernel<<<(unsigned)((F_N + 255) / 256), 256, 0, stream>>>(gbuf);

  // --- 4 directional scans ---
  hprop_kernel<false><<<256, 64, 0, stream>>>(fbuf, gbuf, 0,   o0);  // left->right
  hprop_kernel<true ><<<256, 64, 0, stream>>>(fbuf, gbuf, 96,  o1);  // right->left
  vprop_kernel<true ><<<256, 128, 0, stream>>>(fbuf, gbuf, 192, o2); // bottom->up
  vprop_kernel<false><<<256, 128, 0, stream>>>(fbuf, gbuf, 288, o3); // up->bottom

  // --- max + bilinear 2x upsample -> f16 (8,32,128,256) ---
  maxup_kernel<<<(unsigned)((OUP_N + 255) / 256), 256, 0, stream>>>(o0, o1, o2, o3, oup16);

  // --- conv3 + ReLU -> f16 (8,64,128,256) ---
  // M = 262144 -> 16384 M-tiles x 1 N-group -> 2048 blocks
  conv_wmma_kernel<3, 3, 1, 1, true, true><<<2048, 256, 0, stream>>>(
      oup16, w3p, h1_16, 8, 32, 128, 256, 64, 128, 256, 64);

  // --- conv4 -> fp32 (8,19,128,256) to d_out ---
  conv_wmma_kernel<3, 3, 1, 1, false, false><<<2048, 256, 0, stream>>>(
      h1_16, w4p, d_out, 8, 64, 128, 256, 19, 128, 256, 64);
}
